// PNN_36069135351858
// MI455X (gfx1250) — matrix operand order repaired and checked by reading, not compiled, not run on hardware
//
#include <hip/hip_runtime.h>
#include <hip/hip_bf16.h>

typedef __attribute__((ext_vector_type(16))) __bf16 v16bf;
typedef __attribute__((ext_vector_type(8)))  __bf16 v8bf;
typedef __attribute__((ext_vector_type(8)))  float  v8f;
typedef __attribute__((ext_vector_type(4)))  unsigned int uint4v;
typedef __attribute__((ext_vector_type(8)))  unsigned int uint8v;

#define BDIM   8192
#define DDIM   1024
#define ODIM   1024
#define MDIST  4
#define NDIM   (ODIM * MDIST)   // 4096
#define TILE   128              // block output tile (rows and cols)
#define WTILE  64               // per-wave output tile (rows and cols)
#define KC     32
#define KSTEPS (DDIM / KC)      // 32
#define LDSTR  40               // LDS row stride in halves (80B) via TDM padding

// fp32 -> bf16 round-to-nearest-even
static __device__ __forceinline__ unsigned short f2bf(float f) {
    union { float f; unsigned u; } x; x.f = f;
    unsigned r = x.u + 0x7FFFu + ((x.u >> 16) & 1u);
    return (unsigned short)(r >> 16);
}

static __device__ __forceinline__ v16bf load_frag(const unsigned short* p0,
                                                  const unsigned short* p1) {
    union { v16bf v; v8bf h[2]; } u;
    u.h[0] = *(const v8bf*)p0;
    u.h[1] = *(const v8bf*)p1;
    return u.v;
}

// ---- TDM: DMA one 128-row x 32-half tile (row stride 1024 halves in memory)
// into LDS at lds_off, padding +16B after every 64B so LDS rows land on an
// 80B (LDSTR=40 half) stride.  D# built per CDNA5 ISA 08_async_tensor §8.
static __device__ __forceinline__ void tdm_load_tile(unsigned lds_off,
                                                     unsigned long long gaddr) {
    uint4v g0;
    g0[0] = 1u;                                           // count=1 (valid), user mode
    g0[1] = lds_off;                                      // lds_addr (bytes)
    g0[2] = (unsigned)(gaddr & 0xFFFFFFFFu);              // global_addr[31:0]
    g0[3] = (unsigned)((gaddr >> 32) & 0x01FFFFFFu)       // global_addr[56:32]
          | (2u << 30);                                   // type=2 ("image")
    uint8v g1;
    g1[0] = (1u << 16)                                    // data_size = 2 bytes
          | (1u << 20)                                    // pad_enable
          | (3u << 22)                                    // pad_interval: 16 DWORDs (64B)
          | (3u << 25);                                   // pad_amount: 4 DWORDs (16B)
    g1[1] = ((unsigned)KC   & 0xFFFFu) << 16;             // tensor_dim0[15:0] = 32
    g1[2] = ((unsigned)TILE & 0xFFFFu) << 16;             // dim0[31:16]=0, tensor_dim1 = 128
    g1[3] = ((unsigned)KC   & 0xFFFFu) << 16;             // dim1[31:16]=0, tile_dim0 = 32
    g1[4] = (unsigned)TILE;                               // tile_dim1 = 128, tile_dim2 = 0
    g1[5] = (unsigned)DDIM;                               // tensor_dim0_stride = 1024 elems
    g1[6] = 0u;                                           // stride[47:32], dim1_stride lo
    g1[7] = 0u;                                           // dim1_stride hi (unused, 2D)
    uint4v gz = {0u, 0u, 0u, 0u};                         // groups 2/3 unused (2D tile)
    asm volatile("tensor_load_to_lds %0, %1, %2, %3"
                 :: "s"(g0), "s"(g1), "s"(gz), "s"(gz) : "memory");
}

// -------- Kernel 1: one-shot fp32->bf16 conversion + row squared-norms --------
__global__ __launch_bounds__(256) void pnn_prep_kernel(const float* __restrict__ x,
                                                       const float* __restrict__ cen,
                                                       unsigned short* __restrict__ xbf,
                                                       unsigned short* __restrict__ cbf,
                                                       float* __restrict__ norms) {
    const int lane = threadIdx.x & 31;
    const int wid  = blockIdx.x * (blockDim.x >> 5) + (threadIdx.x >> 5);
    if (wid >= BDIM + NDIM) return;
    const bool isx = wid < BDIM;
    const float* src = isx ? (x + (size_t)wid * DDIM)
                           : (cen + (size_t)(wid - BDIM) * DDIM);
    unsigned short* dst = isx ? (xbf + (size_t)wid * DDIM)
                              : (cbf + (size_t)(wid - BDIM) * DDIM);
    float acc = 0.0f;
    #pragma unroll 4
    for (int i = lane; i < DDIM; i += 32) {
        float v = src[i];
        acc += v * v;
        dst[i] = f2bf(v);
    }
    #pragma unroll
    for (int m = 16; m >= 1; m >>= 1) acc += __shfl_xor(acc, m, 32);
    if (lane == 0) norms[wid] = acc;
}

// -------- Kernel 2: TDM-fed, double-buffered bf16 WMMA GEMM + fused epilogue --
// 4 waves, each owns a 64x64 output tile (16 WMMA accumulators, 4x A/B reuse).
__global__ __launch_bounds__(128) void pnn_main_kernel(const unsigned short* __restrict__ xbf,
                                                       const unsigned short* __restrict__ cbf,
                                                       const float* __restrict__ norms,
                                                       float* __restrict__ out) {
    __shared__ __align__(16) unsigned short sA[2][TILE * LDSTR];
    __shared__ __align__(16) unsigned short sB[2][TILE * LDSTR];
    __shared__ float sx2[TILE];
    __shared__ float sc2[TILE];

    const int tid  = threadIdx.x;
    const int lane = tid & 31;
    const int wid  = tid >> 5;
    const int wr   = wid >> 1;         // wave row (0..1): 64 b-rows
    const int wc   = wid & 1;          // wave col (0..1): 64 n-cols
    const int bB   = blockIdx.y * TILE;
    const int bN   = blockIdx.x * TILE;

    sx2[tid] = norms[bB + tid];        // 128 threads, 128 entries each
    sc2[tid] = norms[BDIM + bN + tid];

    // LDS byte offsets of the staging buffers (low 32 bits of flat LDS address)
    const unsigned ldsA0 = (unsigned)(unsigned long long)(void*)&sA[0][0];
    const unsigned ldsA1 = (unsigned)(unsigned long long)(void*)&sA[1][0];
    const unsigned ldsB0 = (unsigned)(unsigned long long)(void*)&sB[0][0];
    const unsigned ldsB1 = (unsigned)(unsigned long long)(void*)&sB[1][0];
    const unsigned long long gA = (unsigned long long)(xbf + (size_t)bB * DDIM);
    const unsigned long long gC = (unsigned long long)(cbf + (size_t)bN * DDIM);

    v8f acc[4][4];
    #pragma unroll
    for (int i = 0; i < 4; ++i)
        #pragma unroll
        for (int j = 0; j < 4; ++j) acc[i][j] = (v8f){};

    const int lh = lane >> 4;   // half-wave id (0/1)
    const int ll = lane & 15;
    const int ka = 8 * lh;      // A frag: K chunks at ka and ka+16
    const int kb = 16 * lh;     // unused: B frag reads K at ka and ka+16, like A

    // prologue: wave0 kicks TDM for K-tile 0 into buffer 0
    if (wid == 0) {
        tdm_load_tile(ldsA0, gA);
        tdm_load_tile(ldsB0, gC);
    }

    for (int it = 0; it < KSTEPS; ++it) {
        if (wid == 0) __builtin_amdgcn_s_wait_tensorcnt(0);  // current buffer filled
        __syncthreads();
        if (wid == 0 && it + 1 < KSTEPS) {                   // prefetch next K-tile
            const unsigned long long koff = (unsigned long long)(it + 1) * (KC * 2);
            if ((it + 1) & 1) { tdm_load_tile(ldsA1, gA + koff); tdm_load_tile(ldsB1, gC + koff); }
            else              { tdm_load_tile(ldsA0, gA + koff); tdm_load_tile(ldsB0, gC + koff); }
        }
        const unsigned short* bufA = (it & 1) ? &sA[1][0] : &sA[0][0];
        const unsigned short* bufB = (it & 1) ? &sB[1][0] : &sB[0][0];

        // B fragments (16-bit B 32x16): lane = N column, K chunks at ka and ka+16 (same as A).
        v16bf bf[4];
        #pragma unroll
        for (int fc = 0; fc < 4; ++fc) {
            const unsigned short* br = bufB + (WTILE * wc + 16 * fc + ll) * LDSTR;
            bf[fc] = load_frag(br + ka, br + ka + 16);
        }
        // Stream A fragments (16-bit A 16x32): lanes 0-15 hold K 0..7 & 16..23,
        // lanes 16-31 hold K 8..15 & 24..31; 4 WMMAs per A frag.
        #pragma unroll
        for (int fr = 0; fr < 4; ++fr) {
            const unsigned short* ar = bufA + (WTILE * wr + 16 * fr + ll) * LDSTR;
            v16bf a = load_frag(ar + ka, ar + ka + 16);
            #pragma unroll
            for (int fc = 0; fc < 4; ++fc)
                acc[fr][fc] = __builtin_amdgcn_wmma_f32_16x16x32_bf16(
                    false, a, false, bf[fc], (short)0, acc[fr][fc], false, false);
        }
    }

    // ---- fused epilogue: d = x2 - 2*xc + c2 ; p = exp(-2d); combine over M=4 ----
    // C frag: VGPR r -> row (r + 8*lh), column = ll.  The 4 mixture components
    // of one class live in 4 adjacent lanes -> butterfly shfl_xor 1,2.
    #pragma unroll
    for (int fr = 0; fr < 4; ++fr) {
        #pragma unroll
        for (int fc = 0; fc < 4; ++fc) {
            const int rb  = WTILE * wr + 16 * fr + 8 * lh;  // local b-row base
            const int nb  = WTILE * wc + 16 * fc + ll;      // local n-col
            const float c2v = sc2[nb];
            #pragma unroll
            for (int r = 0; r < 8; ++r) {
                float dd = sx2[rb + r] - 2.0f * acc[fr][fc][r] + c2v;
                float p  = __expf(-2.0f * dd);           // sigma = 0.5
                float s  = p + __shfl_xor(p, 1, 32);
                s       += __shfl_xor(s, 2, 32);
                float mx = fmaxf(p, __shfl_xor(p, 1, 32));
                mx       = fmaxf(mx, __shfl_xor(mx, 2, 32));
                if ((lane & 3) == 0) {
                    int gb = bB + rb + r;
                    int go = (bN + nb) >> 2;             // class index
                    out[(size_t)gb * ODIM + go] = s / (s + (float)MDIST - (float)MDIST * mx);
                }
            }
        }
    }
}

extern "C" void kernel_launch(void* const* d_in, const int* in_sizes, int n_in,
                              void* d_out, int out_size, void* d_ws, size_t ws_size,
                              hipStream_t stream) {
    const float* x   = (const float*)d_in[0];   // [8192, 1024]
    const float* cen = (const float*)d_in[1];   // [1024, 4, 1024]
    float*       out = (float*)d_out;           // [8192, 1024]

    // workspace layout: bf16 x copy | bf16 centers copy | norms (x2 ++ c2)
    unsigned short* xbf   = (unsigned short*)d_ws;                  // 16 MB
    unsigned short* cbf   = xbf + (size_t)BDIM * DDIM;              //  8 MB
    float*          norms = (float*)(cbf + (size_t)NDIM * DDIM);    // 48 KB

    // 1) convert to bf16 once + squared norms (one wave per row)
    {
        int rows   = BDIM + NDIM;
        int blocks = (rows + 7) / 8;
        pnn_prep_kernel<<<blocks, 256, 0, stream>>>(x, cen, xbf, cbf, norms);
    }
    // 2) TDM double-buffered WMMA GEMM + gaussian-mixture combine
    {
        dim3 grid(NDIM / TILE, BDIM / TILE);    // (32, 64)
        pnn_main_kernel<<<grid, 128, 0, stream>>>(xbf, cbf, norms, out);
    }
}
